// MultiHeadedAttention_274877907593
// MI455X (gfx1250) — matrix operand order repaired and checked by reading, not compiled, not run on hardware
//
#include <hip/hip_runtime.h>

#define BB 4
#define DMODEL 512
#define NN 2048
#define HEADS 8
#define HDIM 64
#define LOG2E 1.44269504088896340736f

typedef __attribute__((ext_vector_type(16))) __bf16        v16bf;
typedef __attribute__((ext_vector_type(8)))  unsigned int  v8u;
typedef __attribute__((ext_vector_type(8)))  float         v8f;

__device__ __forceinline__ v8f wmma_bf(v16bf a, v16bf b, v8f c) {
    return __builtin_amdgcn_wmma_f32_16x16x32_bf16(
        false, a, false, b, (short)0, c, false, false);
}
__device__ __forceinline__ v8f wmma_u(v8u a, v8u b, v8f c) {
    return wmma_bf(__builtin_bit_cast(v16bf, a), __builtin_bit_cast(v16bf, b), c);
}

// 8 consecutive u32 (16 bf16): one B-fragment (half-select applied by caller)
__device__ __forceinline__ v8u ld8(const unsigned int* p) {
    const uint4* q = (const uint4*)p;
    uint4 a = q[0], b = q[1];
    v8u r = {a.x, a.y, a.z, a.w, b.x, b.y, b.z, b.w};
    return r;
}
// A-fragment from a 64-bf16 row: u32 index j(0..7) -> k*16 + lh*8 + j
__device__ __forceinline__ v8u ldA(const unsigned int* row, int k, int lh) {
    const uint4* q = (const uint4*)row;
    uint4 a = q[k * 4 + lh * 2];
    uint4 b = q[k * 4 + lh * 2 + 1];
    v8u r = {a.x, a.y, a.z, a.w, b.x, b.y, b.z, b.w};
    return r;
}

// ---------------------------------------------------------------------------
// Projection GEMM: out[o,n] = W[o,:] @ X[b,:,n] + bias[o], cast to bf16.
// Each wave computes a 16x64 tile (A fragment reused across 4 n-subtiles).
// Store layout via strides:  addr = bh*NN*HDIM + d*strideD + n*strideN
//   qT/kT [b,h][n][d]: strideD=1,  strideN=HDIM
//   v     [b,h][d][n]: strideD=NN, strideN=1
// ---------------------------------------------------------------------------
__global__ __launch_bounds__(128) void proj_kernel(
    const float* __restrict__ X, const float* __restrict__ W,
    const float* __restrict__ bias, __bf16* __restrict__ out,
    int strideD, int strideN, float scale)
{
    const int w    = threadIdx.x >> 5;
    const int lane = threadIdx.x & 31;
    const int lr   = lane & 15;
    const int lh   = lane >> 4;

    long tile = (long)blockIdx.x * 4 + w;
    const int tn = NN / 64, to = DMODEL / 16;
    int nt = (int)(tile % tn);
    int ot = (int)((tile / tn) % to);
    int b  = (int)(tile / ((long)tn * to));
    int nb = nt * 64, o0 = ot * 16;

    v8f zero = {};
    v8f acc[4];
    #pragma unroll
    for (int t = 0; t < 4; ++t) acc[t] = zero;

    for (int k0 = 0; k0 < DMODEL; k0 += 32) {
        // A fragment: W rows o0..o0+15, K = k0..k0+31, native pk cvt f32->bf16
        v16bf a;
        const float* wrow = W + (long)(o0 + lr) * DMODEL + k0;
        #pragma unroll
        for (int j = 0; j < 8; ++j) {
            int kk = lh * 16 + j * 2;
            a[2 * j]     = (__bf16)wrow[kk];
            a[2 * j + 1] = (__bf16)wrow[kk + 1];
        }
        // 4 B fragments: X[b][k0+K][n], n = nb + t*16 + lr, K = lh*16 + e
        const float* xcol = X + ((long)b * DMODEL + k0 + lh * 16) * NN + nb + lr;
        #pragma unroll
        for (int t = 0; t < 4; ++t) {
            v16bf bm;
            #pragma unroll
            for (int e = 0; e < 16; ++e) bm[e] = (__bf16)xcol[(long)e * NN + t * 16];
            acc[t] = wmma_bf(a, bm, acc[t]);
        }
    }
    // straight-line store (no mode branches: strides select the layout)
    float bv[8];
    #pragma unroll
    for (int r = 0; r < 8; ++r) bv[r] = bias[o0 + r + lh * 8];
    #pragma unroll
    for (int t = 0; t < 4; ++t)
        #pragma unroll
        for (int r = 0; r < 8; ++r) {
            int ch = o0 + r + lh * 8;            // output channel c = d*8 + h
            int n  = nb + t * 16 + lr;
            int hh = ch & 7, d = ch >> 3;
            float val = (acc[t][r] + bv[r]) * scale;
            long addr = ((long)b * HEADS + hh) * (NN * HDIM)
                      + (long)d * strideD + (long)n * strideN;
            out[addr] = (__bf16)val;
        }
}

// ---------------------------------------------------------------------------
// Flash attention: 4 waves/block; wave owns 16 query rows of a 64-row n-tile.
// Scores arrive pre-scaled by log2(e)/sqrt(hd) -> softmax runs in base 2.
// ---------------------------------------------------------------------------
__global__ __launch_bounds__(128) void attn_kernel(
    const __bf16* __restrict__ qT,   // [b,h][n][d] bf16 (pre-scaled)
    const __bf16* __restrict__ kT,   // [b,h][m][d] bf16
    const __bf16* __restrict__ vS,   // [b,h][d][m] bf16
    __bf16* __restrict__ xT)         // [b,h][n][d] bf16
{
    __shared__ __align__(16) __bf16 Plds[4][16][64];

    const int w    = threadIdx.x >> 5;
    const int lane = threadIdx.x & 31;
    const int lr   = lane & 15;
    const int lh   = lane >> 4;

    const int nblks = NN / 64;
    const int nblk  = blockIdx.x % nblks;
    const int bh    = blockIdx.x / nblks;       // b*HEADS + h
    const int n0    = nblk * 64 + w * 16;

    // Q A-fragments: loaded once, reused for every m-chunk (K = d = 0..63)
    const unsigned int* qrow =
        (const unsigned int*)(qT + ((long)bh * NN + n0 + lr) * HDIM);
    v8u aq0 = ldA(qrow, 0, lh);
    v8u aq1 = ldA(qrow, 1, lh);

    float rm[8], rs[8];
    #pragma unroll
    for (int r = 0; r < 8; ++r) { rm[r] = -1e30f; rs[r] = 0.0f; }
    v8f zero = {};
    v8f oacc[4];
    #pragma unroll
    for (int t = 0; t < 4; ++t) oacc[t] = zero;

    for (int m0 = 0; m0 < NN; m0 += 64) {
        // prefetch next K/V chunk toward the caches while computing this one
        if (m0 + 64 < NN) {
            __builtin_prefetch(kT + ((long)bh * NN + m0 + 64 + lr) * HDIM, 0, 1);
            __builtin_prefetch(vS + ((long)bh * HDIM + lr) * NN + m0 + 64, 0, 1);
        }
        // ---- scores S = Q^T K for this 16x64 chunk: 8 WMMA ----
        v8f s[4];
        #pragma unroll
        for (int t = 0; t < 4; ++t) {
            v8f acc = zero;
            #pragma unroll
            for (int k = 0; k < 2; ++k) {
                const unsigned int* krow = (const unsigned int*)
                    (kT + ((long)bh * NN + m0 + t * 16 + lr) * HDIM + k * 32 + lh * 16);
                acc = wmma_u(k ? aq1 : aq0, ld8(krow), acc);
            }
            s[t] = acc;
        }
        // ---- online softmax (base 2): row max over 64 cols ----
        float cmax[8];
        #pragma unroll
        for (int r = 0; r < 8; ++r) {
            float mx = fmaxf(fmaxf(s[0][r], s[1][r]), fmaxf(s[2][r], s[3][r]));
            #pragma unroll
            for (int msk = 1; msk < 16; msk <<= 1)
                mx = fmaxf(mx, __shfl_xor(mx, msk, 32));
            cmax[r] = mx;
        }
        float alpha[8];
        #pragma unroll
        for (int r = 0; r < 8; ++r) {
            float nm = fmaxf(rm[r], cmax[r]);
            alpha[r] = exp2f(rm[r] - nm);
            rm[r] = nm;
        }
        #pragma unroll
        for (int t = 0; t < 4; ++t)
            #pragma unroll
            for (int r = 0; r < 8; ++r) oacc[t][r] *= alpha[r];

        float csum[8];
        #pragma unroll
        for (int r = 0; r < 8; ++r) csum[r] = 0.0f;
        #pragma unroll
        for (int t = 0; t < 4; ++t)
            #pragma unroll
            for (int r = 0; r < 8; ++r) {
                float p = exp2f(s[t][r] - rm[r]);
                csum[r] += p;
                Plds[w][r + lh * 8][t * 16 + lr] = (__bf16)p;   // D-layout -> LDS
            }
        #pragma unroll
        for (int r = 0; r < 8; ++r) {
            float cs = csum[r];
            #pragma unroll
            for (int msk = 1; msk < 16; msk <<= 1) cs += __shfl_xor(cs, msk, 32);
            rs[r] = rs[r] * alpha[r] + cs;
        }
        asm volatile("s_wait_dscnt 0" ::: "memory");   // LDS store->load fence

        // ---- reload P as A-fragments (K = m-local 0..63) ----
        const unsigned int* prow = (const unsigned int*)(&Plds[w][lr][0]);
        v8u ap0 = ldA(prow, 0, lh);
        v8u ap1 = ldA(prow, 1, lh);

        // ---- O += P @ V^T : 8 WMMA ----
        #pragma unroll
        for (int td = 0; td < 4; ++td) {
            #pragma unroll
            for (int k = 0; k < 2; ++k) {
                const unsigned int* vrow = (const unsigned int*)
                    (vS + ((long)bh * HDIM + td * 16 + lr) * NN + m0 + k * 32 + lh * 16);
                oacc[td] = wmma_u(k ? ap1 : ap0, ld8(vrow), oacc[td]);
            }
        }
    }
    // ---- normalize and write xT[b,h][n][d] ----
    #pragma unroll
    for (int td = 0; td < 4; ++td)
        #pragma unroll
        for (int r = 0; r < 8; ++r) {
            int n = n0 + r + lh * 8;
            int d = td * 16 + lr;
            float val = oacc[td][r] / rs[r];
            xT[((long)bh * NN + n) * HDIM + d] = (__bf16)val;
        }
}

// ---------------------------------------------------------------------------
// Output projection: out[b,o,n] = Wm[o,:] @ xhat[:,n] + bm[o]  (fp32 out)
// xhat channel c = d*8 + h gathered from xT[b,h][n][d]; 16x64 tile per wave.
// ---------------------------------------------------------------------------
__global__ __launch_bounds__(128) void outproj_kernel(
    const __bf16* __restrict__ xT, const float* __restrict__ Wm,
    const float* __restrict__ bm, float* __restrict__ out)
{
    const int w    = threadIdx.x >> 5;
    const int lane = threadIdx.x & 31;
    const int lr   = lane & 15;
    const int lh   = lane >> 4;

    long tile = (long)blockIdx.x * 4 + w;
    const int tn = NN / 64, to = DMODEL / 16;
    int nt = (int)(tile % tn);
    int ot = (int)((tile / tn) % to);
    int b  = (int)(tile / ((long)tn * to));
    int nb = nt * 64, o0 = ot * 16;

    v8f zero = {};
    v8f acc[4];
    #pragma unroll
    for (int t = 0; t < 4; ++t) acc[t] = zero;

    for (int k0 = 0; k0 < DMODEL; k0 += 32) {
        v16bf a;
        const float* wrow = Wm + (long)(o0 + lr) * DMODEL + k0;
        #pragma unroll
        for (int j = 0; j < 8; ++j) {
            int kk = lh * 16 + j * 2;
            a[2 * j]     = (__bf16)wrow[kk];
            a[2 * j + 1] = (__bf16)wrow[kk + 1];
        }
        #pragma unroll
        for (int t = 0; t < 4; ++t) {
            v16bf bmm;
            int n = nb + t * 16 + lr;
            #pragma unroll
            for (int e = 0; e < 16; ++e) {
                int c = k0 + lh * 16 + e;
                int hh = c & 7, d = c >> 3;
                bmm[e] = xT[(((long)b * HEADS + hh) * NN + n) * HDIM + d];
            }
            acc[t] = wmma_bf(a, bmm, acc[t]);
        }
    }
    float bv[8];
    #pragma unroll
    for (int r = 0; r < 8; ++r) bv[r] = bm[o0 + r + lh * 8];
    #pragma unroll
    for (int t = 0; t < 4; ++t)
        #pragma unroll
        for (int r = 0; r < 8; ++r) {
            int o = o0 + r + lh * 8;
            int n = nb + t * 16 + lr;
            out[((long)b * DMODEL + o) * NN + n] = acc[t][r] + bv[r];
        }
}

extern "C" void kernel_launch(void* const* d_in, const int* in_sizes, int n_in,
                              void* d_out, int out_size, void* d_ws, size_t ws_size,
                              hipStream_t stream)
{
    const float* query = (const float*)d_in[0];
    const float* key_  = (const float*)d_in[1];
    const float* value = (const float*)d_in[2];
    const float* Wq = (const float*)d_in[3];
    const float* bq = (const float*)d_in[4];
    const float* Wk = (const float*)d_in[5];
    const float* bk = (const float*)d_in[6];
    const float* Wv = (const float*)d_in[7];
    const float* bv = (const float*)d_in[8];
    const float* Wm = (const float*)d_in[9];
    const float* bm = (const float*)d_in[10];
    float* out = (float*)d_out;

    size_t elems = (size_t)BB * DMODEL * NN;   // 4 Mi elements each
    __bf16* qT = (__bf16*)d_ws;                // 8 MB
    __bf16* kT = qT + elems;                   // 8 MB
    __bf16* vS = kT + elems;                   // 8 MB
    __bf16* xT = vS + elems;                   // 8 MB

    const int proj_waves = BB * (DMODEL / 16) * (NN / 64);   // 4096
    dim3 pgrid(proj_waves / 4), pblk(128);

    // q scaled by log2(e)/sqrt(64) so flash softmax runs on native exp2
    proj_kernel<<<pgrid, pblk, 0, stream>>>(query, Wq, bq, qT, 1, HDIM, 0.125f * LOG2E);
    proj_kernel<<<pgrid, pblk, 0, stream>>>(key_,  Wk, bk, kT, 1, HDIM, 1.0f);
    proj_kernel<<<pgrid, pblk, 0, stream>>>(value, Wv, bv, vS, NN, 1, 1.0f);

    dim3 agrid(BB * HEADS * (NN / 64)), ablk(128);           // 1024 blocks
    attn_kernel<<<agrid, ablk, 0, stream>>>(qT, kT, vS, xT);

    outproj_kernel<<<pgrid, pblk, 0, stream>>>(xT, Wm, bm, out);
}
